// MHA_32031866094254
// MI455X (gfx1250) — matrix operand order repaired and checked by reading, not compiled, not run on hardware
//
#include <hip/hip_runtime.h>
#include <hip/hip_bf16.h>

typedef __bf16 bf16;
typedef __attribute__((ext_vector_type(16))) __bf16 bf16x16;
typedef __attribute__((ext_vector_type(8)))  __bf16 bf16x8;
typedef __attribute__((ext_vector_type(8)))  float  f32x8;
typedef __attribute__((ext_vector_type(4)))  float  f32x4;

#define WMMA_BF16(A, Bf, Cf) \
    __builtin_amdgcn_wmma_f32_16x16x32_bf16(false, (A), false, (Bf), (short)0, (Cf), false, false)

// ---------------------------------------------------------------------------
// Fragment builders for v_wmma_f32_16x16x32_bf16 (layouts per CDNA5 ISA 7.12.2)
// A: 16x32 (MxK).  lane<16: M=lane, K = k0+[0..7] and k0+[16..23]
//                  lane>=16: M=lane-16, K = k0+[8..15] and k0+[24..31]
__device__ __forceinline__ bf16x16 load_a_frag(const bf16* __restrict__ base, int ld,
                                               int row0, int k0, int lane) {
    int m  = lane & 15;
    int kb = k0 + ((lane & 16) ? 8 : 0);
    const bf16* p = base + (size_t)(row0 + m) * ld + kb;
    bf16x8 lo = *(const bf16x8*)(p);
    bf16x8 hi = *(const bf16x8*)(p + 16);
    bf16x16 r;
#pragma unroll
    for (int i = 0; i < 8; ++i) { r[i] = lo[i]; r[8 + i] = hi[i]; }
    return r;
}

// B: 32x16 (KxN), B[k][n] = W[n0+n][k]  (multiply by W^T, W row-major)
// lane<16: N=lane, K=k0+[0..7],[16..23]; lane>=16: N=lane-16, K=k0+[8..15],[24..31]
__device__ __forceinline__ bf16x16 load_b_frag(const bf16* __restrict__ W, int ld,
                                               int n0, int k0, int lane) {
    int n  = lane & 15;
    int kb = k0 + ((lane & 16) ? 8 : 0);
    const bf16* p = W + (size_t)(n0 + n) * ld + kb;
    bf16x8 lo = *(const bf16x8*)(p);
    bf16x8 hi = *(const bf16x8*)(p + 16);
    bf16x16 r;
#pragma unroll
    for (int i = 0; i < 8; ++i) { r[i] = lo[i]; r[8 + i] = hi[i]; }
    return r;
}

// ---------------------------------------------------------------------------
__global__ void __launch_bounds__(256)
f32_to_bf16_kernel(const float* __restrict__ x, bf16* __restrict__ y, int n) {
    int i = blockIdx.x * 256 + threadIdx.x;
    if (i < n) y[i] = (bf16)x[i];
}

// ---------------------------------------------------------------------------
// Row LayerNorm (C=1024), f32 in -> bf16 out. 256 threads per row, 4 elems/thread.
__global__ void __launch_bounds__(256)
ln_bf16_kernel(const float* __restrict__ x, const float* __restrict__ g,
               const float* __restrict__ beta, bf16* __restrict__ y) {
    const int C = 1024;
    int row = blockIdx.x;
    int t = threadIdx.x, lane = t & 31, wv = t >> 5;
    const float* xr = x + (size_t)row * C;
    f32x4 vals = *(const f32x4*)(xr + t * 4);
    float s = vals[0] + vals[1] + vals[2] + vals[3];
    float s2 = vals[0]*vals[0] + vals[1]*vals[1] + vals[2]*vals[2] + vals[3]*vals[3];
#pragma unroll
    for (int o = 16; o > 0; o >>= 1) { s += __shfl_xor(s, o, 32); s2 += __shfl_xor(s2, o, 32); }
    __shared__ float red[16];
    if (lane == 0) { red[wv] = s; red[8 + wv] = s2; }
    __syncthreads();
    float ts = 0.f, ts2 = 0.f;
#pragma unroll
    for (int i = 0; i < 8; ++i) { ts += red[i]; ts2 += red[8 + i]; }
    float mean = ts * (1.0f / C);
    float var  = ts2 * (1.0f / C) - mean * mean;
    float inv  = rsqrtf(var + 1e-5f);
    bf16* yr = y + (size_t)row * C + t * 4;
#pragma unroll
    for (int j = 0; j < 4; ++j)
        yr[j] = (bf16)((vals[j] - mean) * inv * g[t * 4 + j] + beta[t * 4 + j]);
}

// ---------------------------------------------------------------------------
// Tiled GEMM:  Y[M,N] = X[M,K] @ W[N,K]^T   (bf16 in, f32 accumulate)
// WG = 256 thr (8 waves), output tile 128x128; each wave owns 32x64.
// K staged in 32-deep chunks through double-buffered LDS using the CDNA5
// async-to-LDS engine.  Steady-state loop is branch-free:
//   stage(next) -> s_wait_asynccnt 4 -> barrier -> 8x WMMA -> barrier
// with the last chunk peeled (wait 0).
// mode 0: store bf16 to Ybf.   mode 1: f32 out = acc + bias[col] + idt (proj).
#define XSTR 40          // padded LDS row stride (elems): bank = 20*m mod 64, 16 distinct

__global__ void __launch_bounds__(256)
gemm_tiled_kernel(const bf16* __restrict__ X, const bf16* __restrict__ W,
                  bf16* __restrict__ Ybf, float* __restrict__ Yf,
                  const float* __restrict__ bias, const float* __restrict__ idt,
                  int M, int N, int K, int mode) {
    __shared__ bf16 Xs[2][128 * XSTR];   // 2 x 10 KB
    __shared__ bf16 Ws[2][128 * XSTR];   // 2 x 10 KB

    int t = threadIdx.x, lane = t & 31, wv = t >> 5;
    int row0g = blockIdx.y * 128, col0g = blockIdx.x * 128;
    int rb = wv & 3, cb = wv >> 2;       // wave tile: rows 32*rb, cols 64*cb

    // Stage one 128x32 X tile + 128x32 W tile into LDS buffer `buf`.
    // 512 x 16B segments per tensor tile; each wave issues 4 async b128
    // instructions per chunk -> ASYNCcnt +4.
    auto stage = [&](int buf, int kc) {
#pragma unroll
        for (int j = 0; j < 2; ++j) {
            int s = t + j * 256;
            int row = s >> 2, part = (s & 3) * 8;                  // element offs
            const bf16* gx = X + (size_t)(row0g + row) * K + kc + part;
            unsigned lx = (unsigned)(unsigned long long)&Xs[buf][row * XSTR + part];
            asm volatile("global_load_async_to_lds_b128 %0, %1, off"
                         :: "v"(lx), "v"(gx) : "memory");
            const bf16* gw = W + (size_t)(col0g + row) * K + kc + part;
            unsigned lw = (unsigned)(unsigned long long)&Ws[buf][row * XSTR + part];
            asm volatile("global_load_async_to_lds_b128 %0, %1, off"
                         :: "v"(lw), "v"(gw) : "memory");
        }
    };

    auto lds_a16 = [&](const bf16* base, int r0) -> bf16x16 {
        int m = lane & 15;
        int kb = (lane & 16) ? 8 : 0;
        const bf16* p = base + (r0 + m) * XSTR + kb;
        bf16x8 lo = *(const bf16x8*)p;
        bf16x8 hi = *(const bf16x8*)(p + 16);
        bf16x16 r;
#pragma unroll
        for (int i = 0; i < 8; ++i) { r[i] = lo[i]; r[8 + i] = hi[i]; }
        return r;
    };
    auto lds_b16 = [&](const bf16* base, int n0) -> bf16x16 {
        int n = lane & 15;
        int kb = (lane & 16) ? 8 : 0;
        const bf16* p = base + (n0 + n) * XSTR + kb;
        bf16x8 lo = *(const bf16x8*)p;
        bf16x8 hi = *(const bf16x8*)(p + 16);
        bf16x16 r;
#pragma unroll
        for (int i = 0; i < 8; ++i) { r[i] = lo[i]; r[8 + i] = hi[i]; }
        return r;
    };

    f32x8 acc[2][4] = {};

    auto compute = [&](int buf) {
        const bf16* xb = Xs[buf];
        const bf16* wb = Ws[buf];
        bf16x16 a0 = lds_a16(xb, rb * 32);
        bf16x16 a1 = lds_a16(xb, rb * 32 + 16);
#pragma unroll
        for (int tt = 0; tt < 4; ++tt) {
            bf16x16 b = lds_b16(wb, cb * 64 + tt * 16);
            acc[0][tt] = WMMA_BF16(a0, b, acc[0][tt]);
            acc[1][tt] = WMMA_BF16(a1, b, acc[1][tt]);
        }
    };

    const int nchunks = K >> 5;
    stage(0, 0);
    // steady state: branch-free prefetch pipeline
    for (int c = 0; c < nchunks - 1; ++c) {
        int cur = c & 1;
        stage(cur ^ 1, (c + 1) << 5);                      // prefetch next chunk
        asm volatile("s_wait_asynccnt 0x4" ::: "memory");  // cur chunk landed
        __syncthreads();                                   // publish LDS tile
        compute(cur);
        __syncthreads();                                   // done reading buffer
    }
    // last chunk
    asm volatile("s_wait_asynccnt 0x0" ::: "memory");
    __syncthreads();
    compute((nchunks - 1) & 1);

    int mlo = (lane & 16) ? 8 : 0, n = lane & 15;
#pragma unroll
    for (int i = 0; i < 2; ++i)
#pragma unroll
        for (int tt = 0; tt < 4; ++tt)
#pragma unroll
            for (int r = 0; r < 8; ++r) {
                int row = row0g + rb * 32 + i * 16 + r + mlo;
                int col = col0g + cb * 64 + tt * 16 + n;
                size_t idx = (size_t)row * N + col;
                if (mode == 0) Ybf[idx] = (bf16)acc[i][tt][r];
                else           Yf[idx]  = acc[i][tt][r] + bias[col] + idt[idx];
            }
}

// ---------------------------------------------------------------------------
// Fused attention for one (b, h, 32-query tile):
//   scores = scale*Q K^T + mask  -> LDS (f32, padded stride)
//   softmax in f32 (attn written to d_out)
//   ctx = P @ V  (V head-slice staged transposed in LDS)
#define SSTR 1028               // 1024 + 4 f32 pad -> bank-conflict free
#define VSTR 1032               // 1024 + 8 bf16 pad, rows stay 16B aligned
#define ATTN_SMEM (32 * SSTR * 4 + 64 * VSTR * 2)

__global__ void __launch_bounds__(256)
attn_kernel(const bf16* __restrict__ Q, const bf16* __restrict__ Kmat,
            const bf16* __restrict__ Vmat, const int* __restrict__ mask,
            float* __restrict__ attn_out, bf16* __restrict__ ctx) {
    const int Cc = 1024, Ns = 1024, Nq = 1024, H = 16, hd = 64;
    extern __shared__ char smem[];
    float* S  = (float*)smem;                       // 32 x SSTR f32
    bf16*  Vt = (bf16*)(smem + 32 * SSTR * 4);      // 64 x VSTR bf16 (transposed V)

    int qt = blockIdx.x, h = blockIdx.y, b = blockIdx.z;
    int lane = threadIdx.x & 31, wv = threadIdx.x >> 5;
    int q0 = qt * 32;
    const bf16* Qh = Q    + (size_t)b * Nq * Cc + h * hd;
    const bf16* Kh = Kmat + (size_t)b * Ns * Cc + h * hd;
    const bf16* Vh = Vmat + (size_t)b * Ns * Cc + h * hd;

    // --- stage V head-slice transposed into LDS: Vt[d][key] ---
    for (int it = 0; it < Ns / 32; ++it) {          // 32 keys per iter
        int key = it * 32 + (threadIdx.x >> 3);
        int d0  = (threadIdx.x & 7) * 8;
        bf16x8 vv = *(const bf16x8*)(Vh + (size_t)key * Cc + d0);
#pragma unroll
        for (int j = 0; j < 8; ++j) Vt[(d0 + j) * VSTR + key] = vv[j];
    }

    // --- phase 1: S = scale * Q K^T + (-1000 * mask) ---
    const float scale = 0.125f;                     // 64^-0.5
    int rb = wv & 1, row0 = rb * 16;
    bf16x16 a0 = load_a_frag(Qh, Cc, q0 + row0, 0, lane);
    bf16x16 a1 = load_a_frag(Qh, Cc, q0 + row0, 32, lane);
    for (int kt = (wv >> 1); kt < 64; kt += 4) {
        int key0 = kt * 16;
        f32x8 acc = {};
        bf16x16 b0 = load_b_frag(Kh, Cc, key0, 0, lane);
        acc = WMMA_BF16(a0, b0, acc);
        bf16x16 b1 = load_b_frag(Kh, Cc, key0, 32, lane);
        acc = WMMA_BF16(a1, b1, acc);
        int n = lane & 15, key = key0 + n;
        float madd = (float)mask[b * Ns + key] * -1000.0f;
        int mlo = (lane & 16) ? 8 : 0;
#pragma unroll
        for (int r = 0; r < 8; ++r)
            S[(row0 + r + mlo) * SSTR + key] = acc[r] * scale + madd;
    }
    __syncthreads();

    // --- phase 2: row softmax; write attn (f32) and P back to LDS ---
    size_t attn_base = ((size_t)(b * H + h) * Nq + q0) * Ns;
#pragma unroll 1
    for (int rr = 0; rr < 4; ++rr) {
        int row = wv * 4 + rr;
        float* Sr = S + row * SSTR;
        float vbuf[32];
        float mx = -1e30f;
#pragma unroll
        for (int i = 0; i < 32; ++i) { vbuf[i] = Sr[lane + i * 32]; mx = fmaxf(mx, vbuf[i]); }
#pragma unroll
        for (int o = 16; o > 0; o >>= 1) mx = fmaxf(mx, __shfl_xor(mx, o, 32));
        float sum = 0.f;
#pragma unroll
        for (int i = 0; i < 32; ++i) { vbuf[i] = __expf(vbuf[i] - mx); sum += vbuf[i]; }
#pragma unroll
        for (int o = 16; o > 0; o >>= 1) sum += __shfl_xor(sum, o, 32);
        float inv = 1.0f / sum;
        float* Ar = attn_out + attn_base + (size_t)row * Ns;
#pragma unroll
        for (int i = 0; i < 32; ++i) {
            float p = vbuf[i] * inv;
            Sr[lane + i * 32] = p;
            Ar[lane + i * 32] = p;
        }
    }
    __syncthreads();

    // --- phase 3: ctx(32x64) = P @ Vhead; each wave one 16x16 tile ---
    int rb3 = wv & 1, cb = wv >> 1;                 // cb in 0..3
    f32x8 acc = {};
    int m = lane & 15;
    int arow = rb3 * 16 + m;
    for (int k0 = 0; k0 < Ns; k0 += 32) {
        int kb = k0 + ((lane & 16) ? 8 : 0);
        const float* Sp = S + arow * SSTR + kb;
        bf16x16 a;
#pragma unroll
        for (int i = 0; i < 8; ++i) { a[i] = (bf16)Sp[i]; a[8 + i] = (bf16)Sp[16 + i]; }
        int n = lane & 15;
        int kvb = k0 + ((lane & 16) ? 8 : 0);
        const bf16* vp = Vt + (cb * 16 + n) * VSTR + kvb;
        bf16x8 blo = *(const bf16x8*)vp;
        bf16x8 bhi = *(const bf16x8*)(vp + 16);
        bf16x16 bfrag;
#pragma unroll
        for (int i = 0; i < 8; ++i) { bfrag[i] = blo[i]; bfrag[8 + i] = bhi[i]; }
        acc = WMMA_BF16(a, bfrag, acc);
    }
    int mlo = (lane & 16) ? 8 : 0, nn = lane & 15;
#pragma unroll
    for (int r = 0; r < 8; ++r) {
        int qrow = q0 + rb3 * 16 + r + mlo;
        ctx[((size_t)b * Nq + qrow) * Cc + h * hd + cb * 16 + nn] = (bf16)acc[r];
    }
}

// ---------------------------------------------------------------------------
extern "C" void kernel_launch(void* const* d_in, const int* in_sizes, int n_in,
                              void* d_out, int out_size, void* d_ws, size_t ws_size,
                              hipStream_t stream) {
    const int B = 8, Nq = 1024, Ns = 1024, C = 1024, D = 1024, H = 16;
    const int M = B * Nq;           // 8192 rows for all big GEMMs

    const float* k_in   = (const float*)d_in[0];
    const float* v_in   = (const float*)d_in[1];
    const float* q_in   = (const float*)d_in[2];
    const float* idt    = (const float*)d_in[3];
    const int*   smask  = (const int*)d_in[4];
    const float* ln_q_g = (const float*)d_in[5];
    const float* ln_q_b = (const float*)d_in[6];
    const float* ln_k_g = (const float*)d_in[7];
    const float* ln_k_b = (const float*)d_in[8];
    const float* ln_v_g = (const float*)d_in[9];
    const float* ln_v_b = (const float*)d_in[10];
    const float* qk_w   = (const float*)d_in[11];
    const float* v_w    = (const float*)d_in[12];
    const float* proj_w = (const float*)d_in[13];
    const float* proj_b = (const float*)d_in[14];

    float* out_ptr  = (float*)d_out;
    float* attn_ptr = out_ptr + (size_t)B * Nq * D;   // attn follows output

    // workspace layout (all bf16)
    char* ws = (char*)d_ws;
    const size_t MB = 1ull << 20;
    bf16* qn   = (bf16*)(ws + 0 * MB);      // 16 MB each activation tensor
    bf16* kn   = (bf16*)(ws + 16 * MB);
    bf16* vn   = (bf16*)(ws + 32 * MB);
    bf16* qkwb = (bf16*)(ws + 48 * MB);     // 2 MB each weight
    bf16* vwb  = (bf16*)(ws + 50 * MB);
    bf16* pwb  = (bf16*)(ws + 52 * MB);
    bf16* Qb   = (bf16*)(ws + 54 * MB);
    bf16* Kb   = (bf16*)(ws + 70 * MB);
    bf16* Vb   = (bf16*)(ws + 86 * MB);
    bf16* ctx  = (bf16*)(ws + 102 * MB);    // total 118 MB

    // 1) weights -> bf16
    int wn = C * C;
    f32_to_bf16_kernel<<<(wn + 255) / 256, 256, 0, stream>>>(qk_w, qkwb, wn);
    f32_to_bf16_kernel<<<(wn + 255) / 256, 256, 0, stream>>>(v_w, vwb, wn);
    f32_to_bf16_kernel<<<(wn + 255) / 256, 256, 0, stream>>>(proj_w, pwb, wn);

    // 2) LayerNorms (f32 stats, bf16 out)
    ln_bf16_kernel<<<M, 256, 0, stream>>>(q_in, ln_q_g, ln_q_b, qn);
    ln_bf16_kernel<<<M, 256, 0, stream>>>(k_in, ln_k_g, ln_k_b, kn);
    ln_bf16_kernel<<<M, 256, 0, stream>>>(v_in, ln_v_g, ln_v_b, vn);

    // 3) input projections via async-staged WMMA GEMM
    dim3 ggrid(C / 128, M / 128);
    gemm_tiled_kernel<<<ggrid, 256, 0, stream>>>(qn, qkwb, Qb, nullptr, nullptr, nullptr,
                                                 M, C, C, 0);
    gemm_tiled_kernel<<<ggrid, 256, 0, stream>>>(kn, qkwb, Kb, nullptr, nullptr, nullptr,
                                                 M, C, C, 0);
    gemm_tiled_kernel<<<ggrid, 256, 0, stream>>>(vn, vwb, Vb, nullptr, nullptr, nullptr,
                                                 M, D, D, 0);

    // 4) fused attention (256 KB dynamic LDS per WG)
    hipFuncSetAttribute((const void*)attn_kernel,
                        hipFuncAttributeMaxDynamicSharedMemorySize, ATTN_SMEM);
    dim3 agrid(Nq / 32, H, B);
    attn_kernel<<<agrid, 256, ATTN_SMEM, stream>>>(Qb, Kb, Vb, smask, attn_ptr, ctx);

    // 5) output projection + bias + residual (f32 out)
    gemm_tiled_kernel<<<ggrid, 256, 0, stream>>>(ctx, pwb, nullptr, out_ptr, proj_b, idt,
                                                 M, D, D, 1);
}
